// NonlinAttention_80891414053491
// MI455X (gfx1250) — matrix operand order repaired and checked by reading, not compiled, not run on hardware
//
#include <hip/hip_runtime.h>
#include <hip/hip_bf16.h>

// ---------------------------------------------------------------------------
// NonlinAttention for MI455X (gfx1250), wave32 + WMMA bf16 + async-to-LDS.
//   K0: convert W_in / W_out f32 -> bf16, pre-swizzled fragment-major tiles
//   K1: gather x by reordered_indexes -> bf16, fragment-major A tiles (xg)
//   K2: GEMM1 xg @ W_in (3 fused segments s/v/y), epilogue v*tanh(s);
//       vt written in attention-B fragment order, y in token order
//   K3: per-(head,b,block) 64x64 @ 64x96 attention, epilogue *y -> z
//       (z written in GEMM2-A fragment order)
//   K4: GEMM2 z @ W_out + b_out, scatter rows to out[:, idx, :]
// All GEMM k-step tiles are flat contiguous copies -> async load to LDS,
// double buffered, with s_wait_asynccnt pipelining.
// ---------------------------------------------------------------------------

typedef __attribute__((ext_vector_type(16))) __bf16        v16bf;
typedef __attribute__((ext_vector_type(8)))  float         v8f;
typedef __attribute__((ext_vector_type(8)))  unsigned int  v8u;
typedef int v4i_vs __attribute__((vector_size(16)));       // builtin's pointee type
typedef __attribute__((address_space(1))) v4i_vs* as1_v4i;
typedef __attribute__((address_space(3))) v4i_vs* as3_v4i;

#define BB   4
#define WW_  128
#define CC   512
#define HID  768
#define HD   96
#define NBT  256
#define BS2  64
#define HWTOK 16384
#define MTOT 65536

#if defined(__gfx1250__) && __has_builtin(__builtin_amdgcn_global_load_async_to_lds_b128)
#define HAVE_ASYNC 1
#else
#define HAVE_ASYNC 0
#endif

__device__ __forceinline__ unsigned short f2bf(float f) {
  unsigned int u = __builtin_bit_cast(unsigned int, f);
  unsigned int r = u + 0x7FFFu + ((u >> 16) & 1u);   // RNE
  return (unsigned short)(r >> 16);
}
__device__ __forceinline__ float bf2f(unsigned short h) {
  unsigned int u = ((unsigned int)h) << 16;
  return __builtin_bit_cast(float, u);
}
__device__ __forceinline__ v16bf load_frag(const unsigned short* p) {
  const uint4* q = (const uint4*)p;
  uint4 a = q[0], b = q[1];
  v8u t = {a.x, a.y, a.z, a.w, b.x, b.y, b.z, b.w};
  return __builtin_bit_cast(v16bf, t);
}
__device__ __forceinline__ v8f vzero8() {
  v8f z;
#pragma unroll
  for (int i = 0; i < 8; ++i) z[i] = 0.0f;
  return z;
}
// 16-byte global -> LDS copy; async path bypasses VGPRs (ASYNCcnt-tracked)
__device__ __forceinline__ void cp16(unsigned short* l, const unsigned short* g) {
#if HAVE_ASYNC
  __builtin_amdgcn_global_load_async_to_lds_b128((as1_v4i)g, (as3_v4i)l, 0, 0);
#else
  *(uint4*)l = *(const uint4*)g;
#endif
}
template <int N> __device__ __forceinline__ void wait_async() {
#if HAVE_ASYNC
#if __has_builtin(__builtin_amdgcn_s_wait_asynccnt)
  __builtin_amdgcn_s_wait_asynccnt(N);
#else
  asm volatile("s_wait_asynccnt %0" ::"i"(N) : "memory");
#endif
#endif
}

// fragment-major address helpers ------------------------------------------
// A-side (16x32 bf16 op): within a (mTile,kstep) 2048-elem block:
//   off = ((st*32 + lane)*16 + g*8 + e8), lane = (m&15) + hbit*16
__device__ __forceinline__ size_t a_sw_off(int m_local, int k_local) {
  int st = m_local >> 4, lm = m_local & 15;
  int g = k_local >> 4, hb = (k_local >> 3) & 1, e8 = k_local & 7;
  return (size_t)((st * 32 + (lm + (hb << 4))) * 16 + g * 8 + e8);
}
// B-side (32x16 bf16 op): within a (nTile,kstep) 2048-elem block:
//   off = ((ni*32 + lane)*16 + g*8 + e8), lane = (n&15) + hbit*16
__device__ __forceinline__ size_t b_sw_off(int k_local, int n_local) {
  int ni = n_local >> 4;
  int lane = (n_local & 15) + (((k_local >> 3) & 1) << 4);
  return (size_t)((ni * 32 + lane) * 16 + (k_local >> 4) * 8 + (k_local & 7));
}

// ---------------------------------------------------------------- K0: weights
__global__ __launch_bounds__(256) void cvt_weights_kernel(
    const float* __restrict__ wi, const float* __restrict__ wo,
    unsigned short* __restrict__ wib, unsigned short* __restrict__ wob) {
  const size_t stride = (size_t)gridDim.x * 256 * 4;
  size_t i0 = ((size_t)blockIdx.x * 256 + threadIdx.x) * 4;
  for (size_t j = i0; j < (size_t)CC * 3 * HID; j += stride) {
    float4 f = *(const float4*)(wi + j);
    float fv[4] = {f.x, f.y, f.z, f.w};
    int k = (int)(j / (3 * HID)), col = (int)(j % (3 * HID));
#pragma unroll
    for (int u = 0; u < 4; ++u) {
      int c = col + u;
      int seg = c / HID, jj = c % HID;
      int nT = jj >> 6, nl = jj & 63;
      size_t base = ((size_t)(seg * 12 + nT) * 16 + (k >> 5)) * 2048;
      wib[base + b_sw_off(k & 31, nl)] = f2bf(fv[u]);
    }
  }
  for (size_t j = i0; j < (size_t)HID * CC; j += stride) {
    float4 f = *(const float4*)(wo + j);
    float fv[4] = {f.x, f.y, f.z, f.w};
    int k = (int)(j / CC), col = (int)(j % CC);
#pragma unroll
    for (int u = 0; u < 4; ++u) {
      int c = col + u;
      int nT = c >> 6, nl = c & 63;
      size_t base = ((size_t)(nT * 24 + (k >> 5))) * 2048;
      wob[base + b_sw_off(k & 31, nl)] = f2bf(fv[u]);
    }
  }
}

// ------------------------------------------------------------- K1: gather+cvt
__global__ __launch_bounds__(256) void gather_cvt_kernel(
    const float* __restrict__ x, const int* __restrict__ idx,
    unsigned short* __restrict__ xg) {
  size_t i = ((size_t)blockIdx.x * 256 + threadIdx.x) * 8;  // over 33554432
  int c = (int)(i & (CC - 1));
  size_t row = i >> 9;                 // b*HWTOK + t
  int b = (int)(row >> 14);
  int t = (int)(row & (HWTOK - 1));
  const float* src = x + (((size_t)b << 14) + (size_t)idx[t]) * CC + c;
  float4 f0 = *(const float4*)src;
  float4 f1 = *(const float4*)(src + 4);
  unsigned short h[8] = {f2bf(f0.x), f2bf(f0.y), f2bf(f0.z), f2bf(f0.w),
                         f2bf(f1.x), f2bf(f1.y), f2bf(f1.z), f2bf(f1.w)};
  // fragment-major dest: 8 consecutive k at fixed row == one 16B chunk
  int mT = (int)(row >> 6), ml = (int)(row & 63);
  size_t dst = (((size_t)mT * 16 + (c >> 5)) * 4) * 512 + a_sw_off(ml, c & 31);
  *(uint4*)(xg + dst) = *(const uint4*)h;
}

// -------------------------------------------------------------- K2: GEMM1
__global__ __launch_bounds__(256) void gemm1_kernel(
    const unsigned short* __restrict__ xg, const unsigned short* __restrict__ wi,
    const float* __restrict__ b_in,
    unsigned short* __restrict__ vt, unsigned short* __restrict__ yb) {
  __shared__ __align__(16) unsigned short As[2][4][32][16];     // 2 x 4KB
  __shared__ __align__(16) unsigned short Bs[2][3][4][32][16];  // 2 x 12KB

  const int tid = threadIdx.x, blk = blockIdx.x;
  const int mTile = blk / 12, nTile = blk % 12;
  const int row0 = mTile * 64, col0 = nTile * 64;
  const int lane = tid & 31, w = tid >> 5;
  const int mi0 = (w >> 2) * 2, ni = w & 3;   // wave: 2 m-subtiles x 1 n x 3 seg

  const unsigned short* aSrc = xg + (size_t)mTile * (64 * CC) + tid * 8;

  v8f acc[3][2];
#pragma unroll
  for (int s = 0; s < 3; ++s) {
    acc[s][0] = vzero8();
    acc[s][1] = vzero8();
  }

  auto issue = [&](int ks, int buf) {
    cp16(&As[buf][0][0][0] + tid * 8, aSrc + (size_t)ks * 2048);
#pragma unroll
    for (int s = 0; s < 3; ++s)
      cp16(&Bs[buf][s][0][0][0] + tid * 8,
           wi + ((size_t)(s * 12 + nTile) * 16 + ks) * 2048 + tid * 8);
  };

  issue(0, 0);
  for (int ks = 0; ks < 16; ++ks) {
    const int buf = ks & 1;
    if (ks + 1 < 16) {
      issue(ks + 1, buf ^ 1);
      wait_async<4>();     // in-order completion: releases current group
    } else {
      wait_async<0>();
    }
    __syncthreads();
    v16bf a0 = load_frag(&As[buf][mi0][lane][0]);
    v16bf a1 = load_frag(&As[buf][mi0 + 1][lane][0]);
#pragma unroll
    for (int s = 0; s < 3; ++s) {
      v16bf bfr = load_frag(&Bs[buf][s][ni][lane][0]);
      acc[s][0] = __builtin_amdgcn_wmma_f32_16x16x32_bf16(
          false, a0, false, bfr, (short)0, acc[s][0], false, false);
      acc[s][1] = __builtin_amdgcn_wmma_f32_16x16x32_bf16(
          false, a1, false, bfr, (short)0, acc[s][1], false, false);
    }
    __syncthreads();
  }

  // epilogue: vt = (v+bv)*tanh(s+bs) -> attention-B fragment order; y token order
  const int jcol = col0 + ni * 16 + (lane & 15);
  const float bs = b_in[jcol], bvv = b_in[HID + jcol], by = b_in[2 * HID + jcol];
  const int head = jcol / HD, d = jcol % HD;
#pragma unroll
  for (int a = 0; a < 2; ++a) {
#pragma unroll
    for (int q = 0; q < 8; ++q) {
      const int mloc = (mi0 + a) * 16 + ((lane >> 4) << 3) + q;
      const int gr = row0 + mloc;
      const int b = gr >> 14, r = gr & (HWTOK - 1);
      const int h = r >> 7, ww = r & 127;
      const int nb = ((h >> 3) << 4) + (ww >> 3);
      const int p = ((h & 7) << 3) + (ww & 7);
      const size_t blkbase = (size_t)(b * NBT + nb) * (BS2 * HID);
      const float sv = acc[0][a][q] + bs;
      const float vv = acc[1][a][q] + bvv;
      // vt: per (b,nb) block, [head][kstep(p>>5)][.] B-fragment order
      const size_t vaddr =
          blkbase + (size_t)head * (BS2 * HD) + (size_t)(p >> 5) * 3072 +
          b_sw_off(p & 31, d);
      vt[vaddr] = f2bf(vv * tanhf(sv));
      yb[blkbase + (size_t)p * HID + jcol] = f2bf(acc[2][a][q] + by);
    }
  }
}

// ------------------------------------------------------- K3: block attention
__global__ __launch_bounds__(256) void attn_kernel(
    const float* __restrict__ attw, const unsigned short* __restrict__ vt,
    const unsigned short* __restrict__ yb, unsigned short* __restrict__ zb) {
  __shared__ __align__(16) unsigned short Als[8][32][16];   // [mi*2+ks][lane][e]
  __shared__ __align__(16) unsigned short Bls[12][32][16];  // [ks*6+ni][lane][e]

  const int tid = threadIdx.x, blk = blockIdx.x;
  const int head = blk >> 10, b = (blk >> 8) & 3, n = blk & 255;
  const float* A = attw + (size_t)blk * 4096;
  const size_t blkbase = (size_t)(b * NBT + n) * (BS2 * HID);

  // V tile: flat 12KB copy (already fragment-major in vt) -> async
  {
    const unsigned short* vSrc = vt + blkbase + (size_t)head * (BS2 * HD);
    unsigned short* bFlat = &Bls[0][0][0];
#pragma unroll
    for (int cch = 0; cch < 3; ++cch)
      cp16(bFlat + (tid + cch * 256) * 8, vSrc + (tid + cch * 256) * 8);
  }
  // A: 64x64 f32 -> bf16 fragment-major (overlaps with async V copy)
  {
    const int p = tid >> 2, qb = (tid & 3) * 16;
#pragma unroll
    for (int i = 0; i < 16; i += 4) {
      float4 f = *(const float4*)(A + p * 64 + qb + i);
      float fv[4] = {f.x, f.y, f.z, f.w};
#pragma unroll
      for (int j = 0; j < 4; ++j) {
        int kk = qb + i + j, ks = kk >> 5, kl = kk & 31;
        int g = kl >> 4, hbit = (kl >> 3) & 1, e = g * 8 + (kl & 7);
        Als[((p >> 4) * 2 + ks)][(p & 15) + (hbit << 4)][e] = f2bf(fv[j]);
      }
    }
  }
  wait_async<0>();
  __syncthreads();

  const int lane = tid & 31, w = tid >> 5;
  v8f acc[3];
#pragma unroll
  for (int t = 0; t < 3; ++t) acc[t] = vzero8();
#pragma unroll
  for (int t = 0; t < 3; ++t) {
    const int tau = w * 3 + t, mi = tau / 6, niv = tau % 6;
#pragma unroll
    for (int ks = 0; ks < 2; ++ks) {
      v16bf af = load_frag(&Als[mi * 2 + ks][lane][0]);
      v16bf bfr = load_frag(&Bls[ks * 6 + niv][lane][0]);
      acc[t] = __builtin_amdgcn_wmma_f32_16x16x32_bf16(
          false, af, false, bfr, (short)0, acc[t], false, false);
    }
  }
  // epilogue: z = vo * y -> GEMM2-A fragment order
#pragma unroll
  for (int t = 0; t < 3; ++t) {
    const int tau = w * 3 + t, mi = tau / 6, niv = tau % 6;
    const int d = niv * 16 + (lane & 15);
    const int j = head * HD + d;                 // hid column
#pragma unroll
    for (int q = 0; q < 8; ++q) {
      const int p = mi * 16 + ((lane >> 4) << 3) + q;
      const float yv = bf2f(yb[blkbase + (size_t)p * HID + j]);
      const int token = (b * NBT + n) * BS2 + p;
      const size_t zaddr =
          ((size_t)(token >> 6) * 24 + (j >> 5)) * 2048 + a_sw_off(token & 63, j & 31);
      zb[zaddr] = f2bf(acc[t][q] * yv);
    }
  }
}

// -------------------------------------------------------------- K4: GEMM2
__global__ __launch_bounds__(256) void gemm2_kernel(
    const unsigned short* __restrict__ zb, const unsigned short* __restrict__ wo,
    const float* __restrict__ b_out, const int* __restrict__ idx,
    float* __restrict__ out) {
  __shared__ __align__(16) unsigned short As[2][4][32][16];
  __shared__ __align__(16) unsigned short Bs[2][4][32][16];

  const int tid = threadIdx.x, blk = blockIdx.x;
  const int mTile = blk >> 3, nTile = blk & 7;
  const int row0 = mTile * 64, col0 = nTile * 64;
  const int lane = tid & 31, w = tid >> 5;
  const int mi0 = (w >> 2) * 2, ni = w & 3;

  const unsigned short* aSrc = zb + (size_t)mTile * (24 * 2048) + tid * 8;
  const unsigned short* bSrc = wo + (size_t)nTile * (24 * 2048) + tid * 8;

  v8f acc[2];
  acc[0] = vzero8();
  acc[1] = vzero8();

  auto issue = [&](int ks, int buf) {
    cp16(&As[buf][0][0][0] + tid * 8, aSrc + (size_t)ks * 2048);
    cp16(&Bs[buf][0][0][0] + tid * 8, bSrc + (size_t)ks * 2048);
  };

  issue(0, 0);
  for (int ks = 0; ks < 24; ++ks) {
    const int buf = ks & 1;
    if (ks + 1 < 24) {
      issue(ks + 1, buf ^ 1);
      wait_async<2>();
    } else {
      wait_async<0>();
    }
    __syncthreads();
    v16bf a0 = load_frag(&As[buf][mi0][lane][0]);
    v16bf a1 = load_frag(&As[buf][mi0 + 1][lane][0]);
    v16bf bfr = load_frag(&Bs[buf][ni][lane][0]);
    acc[0] = __builtin_amdgcn_wmma_f32_16x16x32_bf16(
        false, a0, false, bfr, (short)0, acc[0], false, false);
    acc[1] = __builtin_amdgcn_wmma_f32_16x16x32_bf16(
        false, a1, false, bfr, (short)0, acc[1], false, false);
    __syncthreads();
  }

  const int c = col0 + ni * 16 + (lane & 15);
  const float bo = b_out[c];
#pragma unroll
  for (int a = 0; a < 2; ++a) {
#pragma unroll
    for (int q = 0; q < 8; ++q) {
      const int mloc = (mi0 + a) * 16 + ((lane >> 4) << 3) + q;
      const int tk = row0 + mloc;
      const int b = tk >> 14, qq = tk & (HWTOK - 1);
      const int nb = qq >> 6, p = qq & 63;
      const int h = ((nb >> 4) << 3) + (p >> 3);
      const int ww = ((nb & 15) << 3) + (p & 7);
      const int r = h * WW_ + ww;
      out[((size_t)b * HWTOK + idx[r]) * CC + c] = acc[a][q] + bo;
    }
  }
}

// ---------------------------------------------------------------------------
extern "C" void kernel_launch(void* const* d_in, const int* in_sizes, int n_in,
                              void* d_out, int out_size, void* d_ws, size_t ws_size,
                              hipStream_t stream) {
  (void)in_sizes; (void)n_in; (void)out_size; (void)ws_size;
  const float* x    = (const float*)d_in[0];
  const float* attw = (const float*)d_in[1];
  const float* Wi   = (const float*)d_in[2];
  const float* bi   = (const float*)d_in[3];
  const float* Wo   = (const float*)d_in[4];
  const float* bo   = (const float*)d_in[5];
  const int*   ridx = (const int*)d_in[6];
  float* out = (float*)d_out;

  char* ws = (char*)d_ws;
  unsigned short* xg  = (unsigned short*)(ws);                  //  64 MB
  unsigned short* wib = (unsigned short*)(ws + 67108864);       // 2.25 MB
  unsigned short* wob = (unsigned short*)(ws + 69468160);       // 0.75 MB
  unsigned short* vt  = (unsigned short*)(ws + 70254592);       //  96 MB
  unsigned short* yb  = (unsigned short*)(ws + 170917888);      //  96 MB
  unsigned short* zb  = (unsigned short*)(ws + 271581184);      //  96 MB

  cvt_weights_kernel<<<1152, 256, 0, stream>>>(Wi, Wo, wib, wob);
  gather_cvt_kernel<<<16384, 256, 0, stream>>>(x, ridx, xg);
  gemm1_kernel<<<12288, 256, 0, stream>>>(xg, wib, bi, vt, yb);
  attn_kernel<<<8192, 256, 0, stream>>>(attw, vt, yb, zb);
  gemm2_kernel<<<8192, 256, 0, stream>>>(zb, wob, bo, ridx, out);
}
